// GCN_vanilla_5_layers_31593779430029
// MI455X (gfx1250) — matrix operand order repaired and checked by reading, not compiled, not run on hardware
//
#include <hip/hip_runtime.h>
#include <stdint.h>

// ---------- types ----------
typedef __attribute__((ext_vector_type(16))) __bf16          v16bf;
typedef __attribute__((ext_vector_type(8)))  float           v8f;
typedef __attribute__((ext_vector_type(16))) unsigned short  v16us;
typedef __attribute__((ext_vector_type(8)))  unsigned short  v8us;
typedef __attribute__((ext_vector_type(4)))  unsigned short  v4us;

// bf16 <-> f32 via raw bits (round-to-nearest-even on the downconvert)
__device__ __forceinline__ unsigned short f2bfbits(float f) {
  unsigned u = __builtin_bit_cast(unsigned, f);
  unsigned r = u + 0x7FFFu + ((u >> 16) & 1u);
  return (unsigned short)(r >> 16);
}
__device__ __forceinline__ float bfbits2f(unsigned short s) {
  unsigned u = ((unsigned)s) << 16;
  return __builtin_bit_cast(float, u);
}

// ---------- fp32 -> bf16 conversion (activations) ----------
__global__ void k_f32_to_bf16(const float* __restrict__ in,
                              unsigned short* __restrict__ out, size_t n) {
  size_t i = (size_t)blockIdx.x * blockDim.x + threadIdx.x;
  if (i < n) out[i] = f2bfbits(in[i]);
}

// ---------- pack W (fp32, K x out row-major) into WMMA B-fragment layout ----------
// Packed element p = ((tileN*KB + kb)*32 + lane)*16 + i  holds
//   W[(kb*32 + (i>>3)*16 + (lane>>4)*8 + (i&7)) * out + tileN*16 + (lane&15)]
// so each lane's 16-element B fragment is one contiguous 32-byte block.
__global__ void k_pack_w(const float* __restrict__ W,
                         unsigned short* __restrict__ Wp,
                         int KB, int out, size_t n) {
  size_t p = (size_t)blockIdx.x * blockDim.x + threadIdx.x;
  if (p >= n) return;
  int i    = (int)(p & 15);
  int lane = (int)((p >> 4) & 31);
  int blk  = (int)(p >> 9);
  int kb   = blk % KB;
  int tileN = blk / KB;
  int k   = kb * 32 + (i >> 3) * 16 + (lane >> 4) * 8 + (i & 7);
  int col = tileN * 16 + (lane & 15);
  Wp[p] = f2bfbits(W[(size_t)k * out + col]);
}

// ---------- WMMA GEMM: Z[N x out] = A[N x K] @ W[K x out] (packed B) ----------
// One wave computes a 16 x 64 strip (4 N-tiles), reusing one A fragment per K-step.
__global__ void k_gemm_bf16_wmma(const unsigned short* __restrict__ A,
                                 const unsigned short* __restrict__ Bp,
                                 unsigned short* __restrict__ Z,
                                 int rows16, int K, int out) {
  int gtid   = blockIdx.x * blockDim.x + threadIdx.x;
  int wave   = gtid >> 5;
  int lane   = threadIdx.x & 31;
  int tilesG = out >> 6;                 // groups of 4 tiles along N
  int tileM  = wave / tilesG;
  int tileN0 = (wave - tileM * tilesG) << 2;
  if (tileM >= rows16) return;           // wave-uniform: EXEC stays all-ones
  int half = lane >> 4;
  int r    = lane & 15;
  int KB   = K >> 5;

  v8f acc0 = {}, acc1 = {}, acc2 = {}, acc3 = {};
  const unsigned short* arow = A + (size_t)(tileM * 16 + r) * (size_t)K;
  // per-tile packed B base (advances by 32*16 us per kb step)
  const unsigned short* bbase =
      Bp + ((size_t)tileN0 * KB * 32 + (size_t)lane) * 16;
  const size_t btile = (size_t)KB * 32 * 16;   // elements per packed N-tile
  const size_t bstep = (size_t)32 * 16;        // elements per kb step

  for (int kb = 0; kb < KB; ++kb) {
    v8us alo = *(const v8us*)(arow + kb * 32 + half * 8);
    v8us ahi = *(const v8us*)(arow + kb * 32 + 16 + half * 8);
    v16us au = __builtin_shufflevector(alo, ahi, 0, 1, 2, 3, 4, 5, 6, 7,
                                       8, 9, 10, 11, 12, 13, 14, 15);
    v16bf af = __builtin_bit_cast(v16bf, au);
    const unsigned short* bk = bbase + (size_t)kb * bstep;

    v16bf b0 = __builtin_bit_cast(v16bf, *(const v16us*)(bk));
    v16bf b1 = __builtin_bit_cast(v16bf, *(const v16us*)(bk + btile));
    v16bf b2 = __builtin_bit_cast(v16bf, *(const v16us*)(bk + 2 * btile));
    v16bf b3 = __builtin_bit_cast(v16bf, *(const v16us*)(bk + 3 * btile));

    acc0 = __builtin_amdgcn_wmma_f32_16x16x32_bf16(false, af, false, b0, (short)0, acc0, false, false);
    acc1 = __builtin_amdgcn_wmma_f32_16x16x32_bf16(false, af, false, b1, (short)0, acc1, false, false);
    acc2 = __builtin_amdgcn_wmma_f32_16x16x32_bf16(false, af, false, b2, (short)0, acc2, false, false);
    acc3 = __builtin_amdgcn_wmma_f32_16x16x32_bf16(false, af, false, b3, (short)0, acc3, false, false);
  }

  // D layout: VGPR v -> row M = half*8 + v, col = r (per tile)
  unsigned short* zrow =
      Z + (size_t)(tileM * 16 + half * 8) * (size_t)out + tileN0 * 16 + r;
#pragma unroll
  for (int v = 0; v < 8; ++v) {
    unsigned short* zp = zrow + (size_t)v * out;
    zp[0]  = f2bfbits(acc0[v]);
    zp[16] = f2bfbits(acc1[v]);
    zp[32] = f2bfbits(acc2[v]);
    zp[48] = f2bfbits(acc3[v]);
  }
}

// ---------- acc init with broadcast bias (out is power of two) ----------
__global__ void k_init_bias(float* __restrict__ acc, const float* __restrict__ bias,
                            size_t n, int mask) {
  size_t i = (size_t)blockIdx.x * blockDim.x + threadIdx.x;
  if (i < n) acc[i] = bias[i & (size_t)mask];
}

// ---------- edge scatter: acc[dst] += w * Z[src] (4 features / thread) ----------
__global__ void k_spmm_scatter(const unsigned short* __restrict__ Z,
                               const int* __restrict__ src,
                               const int* __restrict__ dst,
                               const float* __restrict__ ew,
                               float* __restrict__ acc,
                               int E, int shift, int out) {
  long long tid = (long long)blockIdx.x * blockDim.x + threadIdx.x;
  int e = (int)(tid >> shift);
  if (e >= E) return;
  int f = ((int)tid & ((1 << shift) - 1)) << 2;

  float w = ew[e];
  const unsigned short* zr = Z + (size_t)src[e] * (size_t)out + f;
  __builtin_prefetch(zr, 0, 1);               // global_prefetch_b8
  v4us z = *(const v4us*)zr;                  // 8B aligned (f multiple of 4 elems)
  float* ar = acc + (size_t)dst[e] * (size_t)out + f;
  unsafeAtomicAdd(ar + 0, w * bfbits2f(z[0]));
  unsafeAtomicAdd(ar + 1, w * bfbits2f(z[1]));
  unsafeAtomicAdd(ar + 2, w * bfbits2f(z[2]));
  unsafeAtomicAdd(ar + 3, w * bfbits2f(z[3]));
}

// ---------- epilogues ----------
__global__ void k_relu_to_bf16(const float* __restrict__ acc,
                               unsigned short* __restrict__ H, size_t n) {
  size_t i = (size_t)blockIdx.x * blockDim.x + threadIdx.x;
  if (i < n) {
    float v = acc[i];
    H[i] = f2bfbits(v > 0.0f ? v : 0.0f);
  }
}
__global__ void k_copy_f32(const float* __restrict__ acc, float* __restrict__ out,
                           size_t n) {
  size_t i = (size_t)blockIdx.x * blockDim.x + threadIdx.x;
  if (i < n) out[i] = acc[i];
}

// ---------- host ----------
static inline unsigned nblk(size_t n, unsigned t) { return (unsigned)((n + t - 1) / t); }

extern "C" void kernel_launch(void* const* d_in, const int* in_sizes, int n_in,
                              void* d_out, int out_size, void* d_ws, size_t ws_size,
                              hipStream_t stream) {
  (void)n_in; (void)out_size; (void)ws_size;
  static const int DIMS[6] = {256, 512, 256, 128, 64, 64};

  const float* x   = (const float*)d_in[0];
  const int*   src = (const int*)d_in[1];
  const int*   dst = (const int*)d_in[2];
  const float* ew  = (const float*)d_in[3];
  const float* W[5]; const float* b[5];
  for (int i = 0; i < 5; ++i) { W[i] = (const float*)d_in[4 + 2 * i];
                                b[i] = (const float*)d_in[5 + 2 * i]; }
  const int N = in_sizes[0] / DIMS[0];
  const int E = in_sizes[1];

  // workspace layout (256B aligned)
  uintptr_t base = (uintptr_t)d_ws;
  auto aln = [](uintptr_t v) { return (v + 255) & ~(uintptr_t)255; };
  const size_t maxElems = (size_t)N * 512;
  unsigned short* Abuf = (unsigned short*)base; base = aln(base + maxElems * 2);
  unsigned short* Zbuf = (unsigned short*)base; base = aln(base + maxElems * 2);
  float*          accb = (float*)base;          base = aln(base + maxElems * 4);
  unsigned short* Wp[5];
  for (int i = 0; i < 5; ++i) {
    Wp[i] = (unsigned short*)base;
    base = aln(base + (size_t)DIMS[i] * DIMS[i + 1] * 2);
  }

  const unsigned T = 256;

  // x -> bf16 activations
  {
    size_t n = (size_t)N * DIMS[0];
    k_f32_to_bf16<<<nblk(n, T), T, 0, stream>>>(x, Abuf, n);
  }
  // weights -> packed bf16 B-fragment layout
  for (int i = 0; i < 5; ++i) {
    size_t n = (size_t)DIMS[i] * DIMS[i + 1];
    k_pack_w<<<nblk(n, T), T, 0, stream>>>(W[i], Wp[i], DIMS[i] >> 5, DIMS[i + 1], n);
  }

  const int rows16 = N / 16;  // N = 100000 = 6250 * 16
  for (int l = 0; l < 5; ++l) {
    const int in_d  = DIMS[l];
    const int out_d = DIMS[l + 1];

    // 1) Z = A @ W  (bf16 WMMA, f32 accumulate), 4 N-tiles per wave
    {
      long long waves   = (long long)rows16 * (out_d >> 6);
      long long threads = waves * 32;
      k_gemm_bf16_wmma<<<nblk((size_t)threads, T), T, 0, stream>>>(
          Abuf, Wp[l], Zbuf, rows16, in_d, out_d);
    }
    // 2) acc = bias (broadcast)
    {
      size_t n = (size_t)N * out_d;
      k_init_bias<<<nblk(n, T), T, 0, stream>>>(accb, b[l], n, out_d - 1);
    }
    // 3) acc[dst] += w * Z[src]
    {
      int shift = __builtin_ctz(out_d >> 2);  // log2(out_d/4)
      long long total = (long long)E << shift;
      k_spmm_scatter<<<nblk((size_t)total, T), T, 0, stream>>>(
          Zbuf, src, dst, ew, accb, E, shift, out_d);
    }
    // 4) epilogue
    {
      size_t n = (size_t)N * out_d;
      if (l < 4)
        k_relu_to_bf16<<<nblk(n, T), T, 0, stream>>>(accb, Abuf, n);
      else
        k_copy_f32<<<nblk(n, T), T, 0, stream>>>(accb, (float*)d_out, n);
    }
  }
}
